// DynamicLogicalNetwork_63806034149860
// MI455X (gfx1250) — compile-verified, matrix operand order checked by reading, not run on hardware
//
#include <hip/hip_runtime.h>
#include <hip/hip_bf16.h>
#include <stdint.h>

// ---------------------------------------------------------------------------
// Problem constants (fixed by the reference): B=8, S=2048, D_IN=1024,
// H1=2048, H2=1024, C=64.
// ---------------------------------------------------------------------------
#define BQ    8
#define SQ    2048
#define DIN   1024
#define H1Q   2048
#define H2Q   1024
#define CQ    64
#define ROWS  (BQ * SQ)                 // 16384

typedef __attribute__((ext_vector_type(16))) __bf16 v16bf;
typedef __attribute__((ext_vector_type(8)))  float  v8f;

// ---------------------------------------------------------------------------
// Elementwise fp32 -> bf16 convert
// ---------------------------------------------------------------------------
__global__ void __launch_bounds__(256) cvt_bf16_kernel(const float* __restrict__ src,
                                                       __bf16* __restrict__ dst, int n) {
    int i = blockIdx.x * 256 + threadIdx.x;
    if (i < n) dst[i] = (__bf16)src[i];
}

// src is [K][N] fp32 row-major, dst is [N][K] bf16 (i.e. W^T) so that GEMM
// B-fragments are contiguous along K.
__global__ void __launch_bounds__(256) transpose_bf16_kernel(const float* __restrict__ src,
                                                             __bf16* __restrict__ dst,
                                                             int K, int N) {
    int i = blockIdx.x * 256 + threadIdx.x;
    if (i >= N * K) return;
    int n = i / K;
    int k = i - n * K;
    dst[i] = (__bf16)src[(size_t)k * N + n];
}

// ---------------------------------------------------------------------------
// Tiled bf16 WMMA GEMM:  C[M,N] = epi( A[M,K] @ BT[N,K]^T )
//   block tile 128x128, BK = 32 (one v_wmma_f32_16x16x32_bf16 K-step)
//   8 waves / block, wave tile 64(M) x 32(N)  -> 4x2 accumulators of v8f
//   Double-buffered LDS (2 x 12KB per operand), register-staged prefetch:
//   global loads for tile k+2 are issued during the store phase of tile k+1,
//   giving one full compute phase of latency hiding; one barrier per K-step.
//   LDS row stride = 48 bf16 (96B): 16B-aligned b128 fragment reads.
// Epilogues (EPI): 0 = bias + ReLU -> f32 | 1 = *scale -> f32
//                  2 = none -> bf16       | 3 = bias -> f32
// FULL: all tile dims exactly divide (no bounds checks anywhere).
// ---------------------------------------------------------------------------
#define LDS_STRIDE_DW 24               // 48 bf16 per row = 24 dwords = 96 bytes
#define LDS_BUF_DW    (128 * LDS_STRIDE_DW)   // 3072 dwords = 12KB per buffer

template<int EPI, bool FULL>
__global__ void __launch_bounds__(256)
gemm_wmma_bf16(const __bf16* __restrict__ A, const __bf16* __restrict__ BT,
               void* __restrict__ C, const float* __restrict__ bias, float scale,
               int M, int N, int K,
               long long strideA, long long strideB, long long strideC) {
    const int tid  = threadIdx.x;
    const int lane = tid & 31;
    const int wave = tid >> 5;
    const int waveM = wave & 1;        // 0..1 -> 64-row slab
    const int waveN = wave >> 1;       // 0..3 -> 32-col slab
    const int lr = lane & 15;          // row/col within 16x16 tile
    const int lh = lane >> 4;          // K-half selector (0: K 0-15, 1: K 16-31)

    const int blockN = blockIdx.x * 128;
    const int blockM = blockIdx.y * 128;
    const int z      = blockIdx.z;

    const __bf16* Ab = A  + (size_t)z * strideA;
    const __bf16* Bb = BT + (size_t)z * strideB;

    __shared__ uint32_t ldsA[2 * LDS_BUF_DW];
    __shared__ uint32_t ldsB[2 * LDS_BUF_DW];

    // ---- per-thread staging coordinates: 128 rows x 4 16B-chunks ----------
    const int row0 = tid >> 2,         cc0 = tid & 3;          // chunks 0..255
    const int row1 = (tid + 256) >> 2, cc1 = tid & 3;          // chunks 256..511
    const bool vA0 = FULL || (blockM + row0) < M;
    const bool vA1 = FULL || (blockM + row1) < M;
    const bool vB0 = FULL || (blockN + row0) < N;
    const bool vB1 = FULL || (blockN + row1) < N;

    const uint4* pA0 = reinterpret_cast<const uint4*>(Ab + (size_t)(blockM + row0) * K + cc0 * 8);
    const uint4* pA1 = reinterpret_cast<const uint4*>(Ab + (size_t)(blockM + row1) * K + cc1 * 8);
    const uint4* pB0 = reinterpret_cast<const uint4*>(Bb + (size_t)(blockN + row0) * K + cc0 * 8);
    const uint4* pB1 = reinterpret_cast<const uint4*>(Bb + (size_t)(blockN + row1) * K + cc1 * 8);

    const uint32_t stA0 = row0 * LDS_STRIDE_DW + cc0 * 4;
    const uint32_t stA1 = row1 * LDS_STRIDE_DW + cc1 * 4;
    const uint32_t stB0 = stA0, stB1 = stA1;

    const uint4 zero4 = make_uint4(0u, 0u, 0u, 0u);

    v8f acc[4][2];
#pragma unroll
    for (int mt = 0; mt < 4; ++mt)
#pragma unroll
        for (int nt = 0; nt < 2; ++nt)
#pragma unroll
            for (int j = 0; j < 8; ++j) acc[mt][nt][j] = 0.0f;

    const int KT = K >> 5;             // K / 32 steps

    // ---- prologue: tile0 -> lds[0]; issue tile1 -> regs -------------------
    uint4 ra0 = vA0 ? pA0[0] : zero4;  pA0 += 4;
    uint4 ra1 = vA1 ? pA1[0] : zero4;  pA1 += 4;
    uint4 rb0 = vB0 ? pB0[0] : zero4;  pB0 += 4;
    uint4 rb1 = vB1 ? pB1[0] : zero4;  pB1 += 4;
    *reinterpret_cast<uint4*>(&ldsA[stA0]) = ra0;
    *reinterpret_cast<uint4*>(&ldsA[stA1]) = ra1;
    *reinterpret_cast<uint4*>(&ldsB[stB0]) = rb0;
    *reinterpret_cast<uint4*>(&ldsB[stB1]) = rb1;
    if (KT > 1) {
        ra0 = vA0 ? pA0[0] : zero4;  pA0 += 4;
        ra1 = vA1 ? pA1[0] : zero4;  pA1 += 4;
        rb0 = vB0 ? pB0[0] : zero4;  pB0 += 4;
        rb1 = vB1 ? pB1[0] : zero4;  pB1 += 4;
    }
    __syncthreads();

    // ---- fragment base offsets within one LDS buffer ----------------------
    uint32_t fA[4], fB[2];
#pragma unroll
    for (int mt = 0; mt < 4; ++mt)
        fA[mt] = (waveM * 64 + mt * 16 + lr) * LDS_STRIDE_DW + lh * 8;
#pragma unroll
    for (int nt = 0; nt < 2; ++nt)
        fB[nt] = (waveN * 32 + nt * 16 + lr) * LDS_STRIDE_DW + lh * 8;

    for (int kt = 0; kt < KT; ++kt) {
        const uint32_t cur = (kt & 1) * LDS_BUF_DW;
        const uint32_t nxt = ((kt + 1) & 1) * LDS_BUF_DW;

        // store tile kt+1 (in regs, loads in flight) into the idle buffer,
        // then issue global loads for tile kt+2.
        if (kt + 1 < KT) {
            *reinterpret_cast<uint4*>(&ldsA[nxt + stA0]) = ra0;
            *reinterpret_cast<uint4*>(&ldsA[nxt + stA1]) = ra1;
            *reinterpret_cast<uint4*>(&ldsB[nxt + stB0]) = rb0;
            *reinterpret_cast<uint4*>(&ldsB[nxt + stB1]) = rb1;
            if (kt + 2 < KT) {
                ra0 = vA0 ? pA0[0] : zero4;  pA0 += 4;
                ra1 = vA1 ? pA1[0] : zero4;  pA1 += 4;
                rb0 = vB0 ? pB0[0] : zero4;  pB0 += 4;
                rb1 = vB1 ? pB1[0] : zero4;  pB1 += 4;
            }
        }

        // ---- fragments: lane holds 16 contiguous bf16 along K -------------
        v16bf afr[4], bfr[2];
#pragma unroll
        for (int mt = 0; mt < 4; ++mt) {
            const uint4* p = reinterpret_cast<const uint4*>(&ldsA[cur + fA[mt]]);
            reinterpret_cast<uint4*>(&afr[mt])[0] = p[0];
            reinterpret_cast<uint4*>(&afr[mt])[1] = p[1];
        }
#pragma unroll
        for (int nt = 0; nt < 2; ++nt) {
            const uint4* p = reinterpret_cast<const uint4*>(&ldsB[cur + fB[nt]]);
            reinterpret_cast<uint4*>(&bfr[nt])[0] = p[0];
            reinterpret_cast<uint4*>(&bfr[nt])[1] = p[1];
        }

#pragma unroll
        for (int mt = 0; mt < 4; ++mt)
#pragma unroll
            for (int nt = 0; nt < 2; ++nt)
                acc[mt][nt] = __builtin_amdgcn_wmma_f32_16x16x32_bf16(
                    false, afr[mt], false, bfr[nt], (short)0, acc[mt][nt],
                    false, false);
        __syncthreads();
    }

    // ---- epilogue ---------------------------------------------------------
    float*  Cf = reinterpret_cast<float*>(C)  + (size_t)z * strideC;
    __bf16* Ch = reinterpret_cast<__bf16*>(C) + (size_t)z * strideC;

#pragma unroll
    for (int mt = 0; mt < 4; ++mt) {
#pragma unroll
        for (int nt = 0; nt < 2; ++nt) {
            int c = blockN + waveN * 32 + nt * 16 + lr;
            if (!FULL && c >= N) continue;
            int rbase = blockM + waveM * 64 + mt * 16 + lh * 8;
            float bv = (EPI == 0 || EPI == 3) ? bias[c] : 0.0f;
#pragma unroll
            for (int j = 0; j < 8; ++j) {
                int r = rbase + j;
                if (!FULL && r >= M) continue;
                float v = acc[mt][nt][j];
                if constexpr (EPI == 0) {
                    v += bv; v = fmaxf(v, 0.0f);
                    Cf[(size_t)r * N + c] = v;
                } else if constexpr (EPI == 1) {
                    Cf[(size_t)r * N + c] = v * scale;
                } else if constexpr (EPI == 2) {
                    Ch[(size_t)r * N + c] = (__bf16)v;
                } else {
                    Cf[(size_t)r * N + c] = v + bv;
                }
            }
        }
    }
}

// ---------------------------------------------------------------------------
// Row LayerNorm: one block per row. Reads f32, writes bf16 (and optionally a
// transposed bf16 copy [b][d][t] for the P@F GEMM's B operand).
// ---------------------------------------------------------------------------
template<bool WRITE_T>
__global__ void __launch_bounds__(256)
ln_kernel(const float* __restrict__ in, const float* __restrict__ g,
          const float* __restrict__ be, __bf16* __restrict__ out,
          __bf16* __restrict__ outT, int H, int S) {
    const int tid = threadIdx.x;
    const int row = blockIdx.x;
    const float* x = in + (size_t)row * H;
    __shared__ float sred[256];

    float s = 0.0f;
    for (int i = tid; i < H; i += 256) s += x[i];
    sred[tid] = s; __syncthreads();
    for (int off = 128; off > 0; off >>= 1) {
        if (tid < off) sred[tid] += sred[tid + off];
        __syncthreads();
    }
    float mu = sred[0] / (float)H;
    __syncthreads();

    float v = 0.0f;
    for (int i = tid; i < H; i += 256) { float d = x[i] - mu; v += d * d; }
    sred[tid] = v; __syncthreads();
    for (int off = 128; off > 0; off >>= 1) {
        if (tid < off) sred[tid] += sred[tid + off];
        __syncthreads();
    }
    float inv = rsqrtf(sred[0] / (float)H + 1e-5f);

    int b = row / S, t = row - b * S;
    for (int i = tid; i < H; i += 256) {
        float y = (x[i] - mu) * inv * g[i] + be[i];
        out[(size_t)row * H + i] = (__bf16)y;
        if constexpr (WRITE_T)
            outT[(size_t)b * H * S + (size_t)i * S + t] = (__bf16)y;
    }
}

// ---------------------------------------------------------------------------
// Row softmax over T elems, fp32 in place (this region IS the
// attention_weights output) + bf16 copy for the P@F GEMM.
// ---------------------------------------------------------------------------
__global__ void __launch_bounds__(256)
softmax_kernel(float* __restrict__ attn, __bf16* __restrict__ pbf, int T) {
    const int tid = threadIdx.x;
    const int row = blockIdx.x;
    float* p = attn + (size_t)row * T;
    __shared__ float sred[256];

    float m = -3.4e38f;
    for (int i = tid; i < T; i += 256) m = fmaxf(m, p[i]);
    sred[tid] = m; __syncthreads();
    for (int off = 128; off > 0; off >>= 1) {
        if (tid < off) sred[tid] = fmaxf(sred[tid], sred[tid + off]);
        __syncthreads();
    }
    float mx = sred[0];
    __syncthreads();

    float s = 0.0f;
    for (int i = tid; i < T; i += 256) { float e = __expf(p[i] - mx); p[i] = e; s += e; }
    sred[tid] = s; __syncthreads();
    for (int off = 128; off > 0; off >>= 1) {
        if (tid < off) sred[tid] += sred[tid + off];
        __syncthreads();
    }
    float inv = 1.0f / sred[0];

    for (int i = tid; i < T; i += 256) {
        float v = p[i] * inv;
        p[i] = v;
        pbf[(size_t)row * T + i] = (__bf16)v;
    }
}

// ---------------------------------------------------------------------------
// Soft AND/OR gate head epilogue, in place on logits [B][S][C].
// ---------------------------------------------------------------------------
__global__ void __launch_bounds__(256)
gating_kernel(float* __restrict__ logits, const float* __restrict__ prior,
              const float* __restrict__ ag, const float* __restrict__ og) {
    int i = blockIdx.x * 256 + threadIdx.x;   // total B*S*C = 1<<20
    int c = i & (CQ - 1);
    int b = i >> 17;                          // S*C = 2^17
    float x  = logits[i];
    float sl = 1.0f / (1.0f + __expf(-x));
    float sp = 1.0f / (1.0f + __expf(-prior[(b << 6) + c]));
    float av = fminf(sl, sp) * ag[c];
    float ov = fmaxf(sl, sp) * og[c];
    logits[i] = __logf((av + ov) * 0.5f + 1e-10f);
}

// ---------------------------------------------------------------------------
// Workspace layout (bytes). Buffers are aggressively reused:
//   tmp  : f32 pre-LN activations (GEMM1 out 128MB, then GEMM2 out 64MB)
//   XBF  : X bf16 (32MB); dead after GEMM1 -> reused for `attended` bf16
//   HBF  : h bf16 (64MB); dead after GEMM2 -> reused for P bf16
// Total ~297 MB.
// ---------------------------------------------------------------------------
#define OFF_TMP   ((size_t)0)
#define OFF_XBF   ((size_t)134217728)                 // 128MB
#define OFF_HBF   (OFF_XBF + (size_t)33554432)        // +32MB
#define OFF_FBF   (OFF_HBF + (size_t)67108864)        // +64MB
#define OFF_FTBF  (OFF_FBF + (size_t)33554432)        // +32MB
#define OFF_W1T   (OFF_FTBF + (size_t)33554432)       // +32MB
#define OFF_W2T   (OFF_W1T + (size_t)4194304)
#define OFF_WHT   (OFF_W2T + (size_t)4194304)

extern "C" void kernel_launch(void* const* d_in, const int* in_sizes, int n_in,
                              void* d_out, int out_size, void* d_ws, size_t ws_size,
                              hipStream_t stream) {
    const float* emb   = (const float*)d_in[0];
    const float* prior = (const float*)d_in[1];
    const float* W1    = (const float*)d_in[2];
    const float* b1    = (const float*)d_in[3];
    const float* g1    = (const float*)d_in[4];
    const float* beta1 = (const float*)d_in[5];
    const float* W2    = (const float*)d_in[6];
    const float* b2    = (const float*)d_in[7];
    const float* g2    = (const float*)d_in[8];
    const float* beta2 = (const float*)d_in[9];
    const float* Wh    = (const float*)d_in[10];
    const float* bh    = (const float*)d_in[11];
    const float* ag    = (const float*)d_in[12];
    const float* og    = (const float*)d_in[13];

    char* ws = (char*)d_ws;
    float*  tmp  = (float*)(ws + OFF_TMP);
    __bf16* Xbf  = (__bf16*)(ws + OFF_XBF);
    __bf16* abf  = Xbf;                        // reuse after GEMM1
    __bf16* hbf  = (__bf16*)(ws + OFF_HBF);
    __bf16* pbf  = hbf;                        // reuse after GEMM2
    __bf16* fbf  = (__bf16*)(ws + OFF_FBF);
    __bf16* FTbf = (__bf16*)(ws + OFF_FTBF);
    __bf16* W1T  = (__bf16*)(ws + OFF_W1T);
    __bf16* W2T  = (__bf16*)(ws + OFF_W2T);
    __bf16* WhT  = (__bf16*)(ws + OFF_WHT);

    float* logits = (float*)d_out;                        // [B,S,C] f32
    float* attn   = (float*)d_out + (size_t)BQ * SQ * CQ; // [B,S,S] f32

    // --- precision conversion (weights pre-transposed to [N][K] bf16) ------
    cvt_bf16_kernel<<<(ROWS * DIN) / 256, 256, 0, stream>>>(emb, Xbf, ROWS * DIN);
    transpose_bf16_kernel<<<(DIN * H1Q) / 256, 256, 0, stream>>>(W1, W1T, DIN, H1Q);
    transpose_bf16_kernel<<<(H1Q * H2Q) / 256, 256, 0, stream>>>(W2, W2T, H1Q, H2Q);
    transpose_bf16_kernel<<<(H2Q * CQ) / 256, 256, 0, stream>>>(Wh, WhT, H2Q, CQ);

    // --- layer 1: h = LN(relu(X @ W1 + b1)) --------------------------------
    {
        dim3 g(H1Q / 128, ROWS / 128, 1);
        gemm_wmma_bf16<0, true><<<g, 256, 0, stream>>>(Xbf, W1T, tmp, b1, 1.0f,
                                                       ROWS, H1Q, DIN, 0, 0, 0);
        ln_kernel<false><<<ROWS, 256, 0, stream>>>(tmp, g1, beta1, hbf, nullptr, H1Q, SQ);
    }
    // --- layer 2: F = LN(relu(h @ W2 + b2)), plus F^T copy -----------------
    {
        dim3 g(H2Q / 128, ROWS / 128, 1);
        gemm_wmma_bf16<0, true><<<g, 256, 0, stream>>>(hbf, W2T, tmp, b2, 1.0f,
                                                       ROWS, H2Q, H1Q, 0, 0, 0);
        ln_kernel<true><<<ROWS, 256, 0, stream>>>(tmp, g2, beta2, fbf, FTbf, H2Q, SQ);
    }
    // --- scores = F @ F^T / sqrt(d), straight into attention output --------
    {
        dim3 g(SQ / 128, SQ / 128, BQ);
        gemm_wmma_bf16<1, true><<<g, 256, 0, stream>>>(fbf, fbf, attn, nullptr, 0.03125f,
                                                       SQ, SQ, H2Q,
                                                       (long long)SQ * H2Q,
                                                       (long long)SQ * H2Q,
                                                       (long long)SQ * SQ);
        softmax_kernel<<<ROWS, 256, 0, stream>>>(attn, pbf, SQ);
    }
    // --- attended = P @ F  (B operand = F^T rows, contiguous along K=t) ----
    {
        dim3 g(H2Q / 128, SQ / 128, BQ);
        gemm_wmma_bf16<2, true><<<g, 256, 0, stream>>>(pbf, FTbf, abf, nullptr, 1.0f,
                                                       SQ, H2Q, SQ,
                                                       (long long)SQ * SQ,
                                                       (long long)H2Q * SQ,
                                                       (long long)SQ * H2Q);
    }
    // --- logits = attended @ Wh + bh, then soft AND/OR gate ----------------
    {
        dim3 g(1, ROWS / 128, 1);
        gemm_wmma_bf16<3, false><<<g, 256, 0, stream>>>(abf, WhT, logits, bh, 1.0f,
                                                        ROWS, CQ, H2Q, 0, 0, 0);
        gating_kernel<<<(BQ * SQ * CQ) / 256, 256, 0, stream>>>(logits, prior, ag, og);
    }
    (void)in_sizes; (void)n_in; (void)out_size; (void)ws_size;
}
